// ComplexBasicMultiheadAttention_13469017440841
// MI455X (gfx1250) — matrix operand order repaired and checked by reading, not compiled, not run on hardware
//
#include <hip/hip_runtime.h>

typedef unsigned short u16;
typedef unsigned int   u32;
typedef unsigned long long u64;
typedef __bf16 bf16_t;
typedef bf16_t v16bf __attribute__((ext_vector_type(16)));
typedef float  v8f   __attribute__((ext_vector_type(8)));
typedef int    v4i   __attribute__((ext_vector_type(4)));
typedef int    v8i   __attribute__((ext_vector_type(8)));

#define S_DIM 1024
#define B_DIM 4
#define E_DIM 1024

__device__ __forceinline__ u16 f2bf(float f) {
    union { float f; u32 u; } v; v.f = f;
    u32 r = (v.u + 0x7FFFu + ((v.u >> 16) & 1u)) >> 16;   // RNE
    return (u16)r;
}
__device__ __forceinline__ u32 pack2(float a, float b) {
    return (u32)f2bf(a) | ((u32)f2bf(b) << 16);
}

union Frag16 { v16bf v; uint4 q[2]; u32 w[8]; u16 u[16]; };
union Out8   { uint4 q; u16 u[8]; };

__device__ __forceinline__ v8f wmma_bf16(v16bf a, v16bf b, v8f c) {
    return __builtin_amdgcn_wmma_f32_16x16x32_bf16(false, a, false, b,
                                                   (short)0, c, false, false);
}

// ---------------------------------------------------------------------------
// Tensor Data Mover: 2D tile global->LDS, 2-group D# (VADDR2/3 = NULL).
// D# group0: count=1 | lds_addr | global_addr | type=2
// D# group1: data_size=2B, pad_enable, pad_interval/amount, dims/strides
// All dims/strides in elements (data_size units).
// ---------------------------------------------------------------------------
__device__ __forceinline__ u32 lds_off(const void* p) {
    return (u32)(u64)(uintptr_t)p;     // LDS aperture: addr[31:0] = LDS offset
}

// flags: row width 32 elems (16 DW) pad 8 elems (4 DW) -> pitch 40
#define TDM_F32PAD ((1u << 16) | (1u << 20) | (3u << 22) | (3u << 25))
// flags: row width 64 elems (32 DW) pad 8 elems (4 DW) -> pitch 72
#define TDM_F64PAD ((1u << 16) | (1u << 20) | (4u << 22) | (3u << 25))

__device__ __forceinline__ void tdm_load_2d(u32 lds_byte, const u16* gptr,
                                            u32 flags, u32 td0, u32 td1,
                                            u32 tile0, u32 tile1, u32 stride0) {
    u64 ga = (u64)(uintptr_t)gptr;
    v4i g0;
    g0[0] = 1;                                          // count = 1
    g0[1] = (int)lds_byte;                              // lds_addr
    g0[2] = (int)(u32)ga;                               // global_addr[31:0]
    g0[3] = (int)(((u32)(ga >> 32) & 0x01FFFFFFu) | 0x80000000u);  // ga[56:32] | type=2
    v8i g1;
    g1[0] = (int)flags;                                 // mask=0, ds=2B, pad cfg
    g1[1] = (int)(td0 << 16);                           // tensor_dim0[15:0]
    g1[2] = (int)((td0 >> 16) | (td1 << 16));           // tensor_dim0[31:16] | tensor_dim1[15:0]
    g1[3] = (int)((td1 >> 16) | (tile0 << 16));         // tensor_dim1[31:16] | tile_dim0
    g1[4] = (int)tile1;                                 // tile_dim1 (tile_dim2=0)
    g1[5] = (int)stride0;                               // tensor_dim0_stride[31:0]
    g1[6] = 0;
    g1[7] = 0;
    asm volatile("tensor_load_to_lds %0, %1" :: "s"(g0), "s"(g1) : "memory");
}

// ---------------------------------------------------------------------------
// Pack [S,B,E] fp32 -> [B,S,E] bf16
// ---------------------------------------------------------------------------
__global__ __launch_bounds__(256) void pack_in_kernel(const float* __restrict__ src,
                                                      u16* __restrict__ dst) {
    int t = blockIdx.x * 256 + threadIdx.x;
    int base = t * 4;                    // flat index into [S,B,E]
    int e  = base & (E_DIM - 1);
    int sb = base >> 10;
    int b  = sb & (B_DIM - 1);
    int s  = sb >> 2;
    float4 x = *(const float4*)(src + base);
    uint2 p; p.x = pack2(x.x, x.y); p.y = pack2(x.z, x.w);
    *(uint2*)(dst + (((size_t)(b * S_DIM + s)) << 10) + e) = p;
}

__global__ __launch_bounds__(256) void pack_w_kernel(const float* __restrict__ src,
                                                     u16* __restrict__ dst) {
    int base = (blockIdx.x * 256 + threadIdx.x) * 4;
    float4 x = *(const float4*)(src + base);
    uint2 p; p.x = pack2(x.x, x.y); p.y = pack2(x.z, x.w);
    *(uint2*)(dst + base) = p;
}

// ---------------------------------------------------------------------------
// Complex GEMM: C = A @ W^T + bias, TDM double-buffered staging.
// MODE 0: bf16 [B,H,S,D] (Q,K) | MODE 1: bf16 [B,H,D,S] (V^T) | MODE 2: fp32 [S,B,E,2]
// 256 threads / 8 waves; tile M=128 (16 rows/wave), N=64 (4 n-tiles)
// ---------------------------------------------------------------------------
template<int MODE>
__global__ __launch_bounds__(256) void cgemm_kernel(
    const u16* __restrict__ Ar, const u16* __restrict__ Ai,
    const u16* __restrict__ Wr, const u16* __restrict__ Wi,
    const float* __restrict__ br, const float* __restrict__ bi,
    u16* __restrict__ Or_, u16* __restrict__ Oi_, float* __restrict__ outF)
{
    constexpr int PITCH = 40;
    __shared__ __align__(16) u16 lAr[2 * 128 * PITCH];
    __shared__ __align__(16) u16 lAi[2 * 128 * PITCH];
    __shared__ __align__(16) u16 lWr[2 * 64 * PITCH];
    __shared__ __align__(16) u16 lWi[2 * 64 * PITCH];

    const int tid  = threadIdx.x;
    const int wv   = tid >> 5;
    const int lane = tid & 31;
    const int hi   = lane >> 4;
    const int ln   = lane & 15;
    const int bm   = blockIdx.x;      // 0..31
    const int bn   = blockIdx.y;      // 0..15

    const size_t aRow = (size_t)bm * 128 * 1024;
    const size_t wRow = (size_t)bn * 64 * 1024;

    auto issue = [&](int buf, int k0) {
        tdm_load_2d(lds_off(lAr) + buf * (128 * PITCH * 2), Ar + aRow + k0,
                    TDM_F32PAD, 1024, 4096, 32, 128, 1024);
        tdm_load_2d(lds_off(lAi) + buf * (128 * PITCH * 2), Ai + aRow + k0,
                    TDM_F32PAD, 1024, 4096, 32, 128, 1024);
        tdm_load_2d(lds_off(lWr) + buf * (64 * PITCH * 2), Wr + wRow + k0,
                    TDM_F32PAD, 1024, 1024, 32, 64, 1024);
        tdm_load_2d(lds_off(lWi) + buf * (64 * PITCH * 2), Wi + wRow + k0,
                    TDM_F32PAD, 1024, 1024, 32, 64, 1024);
    };

    if (wv == 0) issue(0, 0);

    v8f acc_rr[4] = {}, acc_ii[4] = {}, acc_i[4] = {};

    int buf = 0;
    for (int k0 = 0; k0 < 1024; k0 += 32) {
        if (wv == 0) {
            if (k0 + 32 < 1024) {
                issue(buf ^ 1, k0 + 32);                 // prefetch next slab
                __builtin_amdgcn_s_wait_tensorcnt(4);    // current slab done
            } else {
                __builtin_amdgcn_s_wait_tensorcnt(0);
            }
        }
        __syncthreads();

        const u16* pAr = lAr + buf * (128 * PITCH);
        const u16* pAi = lAi + buf * (128 * PITCH);
        const u16* pWr = lWr + buf * (64 * PITCH);
        const u16* pWi = lWi + buf * (64 * PITCH);

        Frag16 a_r, a_i;
        {
            const u16* p = pAr + (wv * 16 + ln) * PITCH;
            a_r.q[0] = *(const uint4*)(p + hi * 8);
            a_r.q[1] = *(const uint4*)(p + 16 + hi * 8);
            p = pAi + (wv * 16 + ln) * PITCH;
            a_i.q[0] = *(const uint4*)(p + hi * 8);
            a_i.q[1] = *(const uint4*)(p + 16 + hi * 8);
        }
        #pragma unroll
        for (int nt = 0; nt < 4; ++nt) {
            Frag16 b_r, b_i;
            const u16* p = pWr + (nt * 16 + ln) * PITCH + hi * 8;
            b_r.q[0] = *(const uint4*)(p);
            b_r.q[1] = *(const uint4*)(p + 16);
            p = pWi + (nt * 16 + ln) * PITCH + hi * 8;
            b_i.q[0] = *(const uint4*)(p);
            b_i.q[1] = *(const uint4*)(p + 16);

            acc_rr[nt] = wmma_bf16(a_r.v, b_r.v, acc_rr[nt]);
            acc_ii[nt] = wmma_bf16(a_i.v, b_i.v, acc_ii[nt]);
            acc_i[nt]  = wmma_bf16(a_r.v, b_i.v,
                          wmma_bf16(a_i.v, b_r.v, acc_i[nt]));
        }
        __syncthreads();
        buf ^= 1;
    }

    #pragma unroll
    for (int nt = 0; nt < 4; ++nt) {
        int gn = bn * 64 + nt * 16 + ln;
        float vbr = br[gn], vbi = bi[gn];
        int h = gn >> 6, d = gn & 63;
        #pragma unroll
        for (int r = 0; r < 8; ++r) {
            int m = bm * 128 + wv * 16 + hi * 8 + r;   // m = b*S + s
            int b = m >> 10, s = m & 1023;
            float cr = acc_rr[nt][r] - acc_ii[nt][r] + vbr;
            float ci = acc_i[nt][r] + vbi;
            if (MODE == 0) {
                size_t o = ((((size_t)(b * 16 + h)) << 10) + s) * 64 + d;
                Or_[o] = f2bf(cr); Oi_[o] = f2bf(ci);
            } else if (MODE == 1) {
                size_t o = (((size_t)(b * 16 + h) * 64 + d) << 10) + s;
                Or_[o] = f2bf(cr); Oi_[o] = f2bf(ci);
            } else {
                size_t o = ((((size_t)(s * 4 + b)) << 10) + gn) * 2;
                *(float2*)(outF + o) = make_float2(cr, ci);
            }
        }
    }
}

// ---------------------------------------------------------------------------
// Flash attention, split softmax. 8 waves/block share one (b,h); each wave
// owns one 16-query tile. K/V 32-key chunks staged once per block via TDM
// (double-buffered) instead of 8x-redundant global reads.
// Q,K: [B,H,S,D] bf16. V: [B,H,D,S] bf16. Out: [B*S, E] bf16.
// ---------------------------------------------------------------------------
__global__ __launch_bounds__(256) void attn_kernel(
    const u16* __restrict__ Qr, const u16* __restrict__ Qi,
    const u16* __restrict__ Kr, const u16* __restrict__ Ki,
    const u16* __restrict__ Vr, const u16* __restrict__ Vi,
    u16* __restrict__ Or_, u16* __restrict__ Oi_)
{
    constexpr int KP = 72;   // K tile pitch (64 + 8 pad)
    constexpr int VP = 40;   // V tile pitch (32 + 8 pad)
    __shared__ __align__(16) u16 sKr[2 * 32 * KP];
    __shared__ __align__(16) u16 sKi[2 * 32 * KP];
    __shared__ __align__(16) u16 sVr[2 * 64 * VP];
    __shared__ __align__(16) u16 sVi[2 * 64 * VP];

    const int tid  = threadIdx.x;
    const int wv   = tid >> 5;
    const int lane = tid & 31;
    const int hi   = lane >> 4;
    const int ln   = lane & 15;
    const int gw   = blockIdx.x * 8 + wv;   // global wave id
    const int q0   = (gw & 63) * 16;        // query tile base
    const int bh   = gw >> 6;               // shared by all 8 waves of block

    const size_t base = (size_t)bh << 16;   // bh * S * D

    auto issue = [&](int buf, int kb) {
        // K chunk: 32 key-rows x 64 d (row-major, stride 64)
        tdm_load_2d(lds_off(sKr) + buf * (32 * KP * 2), Kr + base + (size_t)kb * 64,
                    TDM_F64PAD, 64, 1024, 64, 32, 64);
        tdm_load_2d(lds_off(sKi) + buf * (32 * KP * 2), Ki + base + (size_t)kb * 64,
                    TDM_F64PAD, 64, 1024, 64, 32, 64);
        // V^T chunk: 64 d-rows x 32 k (row stride S=1024)
        tdm_load_2d(lds_off(sVr) + buf * (64 * VP * 2), Vr + base + kb,
                    TDM_F32PAD, 1024, 64, 32, 64, 1024);
        tdm_load_2d(lds_off(sVi) + buf * (64 * VP * 2), Vi + base + kb,
                    TDM_F32PAD, 1024, 64, 32, 64, 1024);
    };

    if (wv == 0) issue(0, 0);

    // Q as B-fragments (B[d,q] = Q[q,d]); loaded once from global.
    Frag16 qbr[2], qbi[2];
    {
        const u16* qr = Qr + base + (size_t)(q0 + ln) * 64;
        const u16* qi = Qi + base + (size_t)(q0 + ln) * 64;
        #pragma unroll
        for (int c = 0; c < 2; ++c) {
            const u16* pr = qr + c * 32 + hi * 16;
            const u16* pi = qi + c * 32 + hi * 16;
            qbr[c].q[0] = *(const uint4*)(pr); qbr[c].q[1] = *(const uint4*)(pr + 8);
            qbi[c].q[0] = *(const uint4*)(pi); qbi[c].q[1] = *(const uint4*)(pi + 8);
        }
    }

    v8f Arr[4] = {}, Aii[4] = {}, Ari[4] = {}, Air[4] = {};
    float m_r = -1e30f, m_i = -1e30f, l_r = 0.f, l_i = 0.f;

    int buf = 0;
    for (int kb = 0; kb < 1024; kb += 32) {
        if (wv == 0) {
            if (kb + 32 < 1024) {
                issue(buf ^ 1, kb + 32);
                __builtin_amdgcn_s_wait_tensorcnt(4);
            } else {
                __builtin_amdgcn_s_wait_tensorcnt(0);
            }
        }
        __syncthreads();

        const u16* kbr = sKr + buf * (32 * KP);
        const u16* kbi = sKi + buf * (32 * KP);
        const u16* vbr = sVr + buf * (64 * VP);
        const u16* vbi = sVi + buf * (64 * VP);

        float p_r[2][8], p_i[2][8];
        // ---- transposed scores St[k,q] for two 16-key tiles ----
        #pragma unroll
        for (int t = 0; t < 2; ++t) {
            const u16* krp = kbr + (t * 16 + ln) * KP;
            const u16* kip = kbi + (t * 16 + ln) * KP;
            Frag16 ka_r[2], ka_i[2];
            #pragma unroll
            for (int c = 0; c < 2; ++c) {
                const u16* pr = krp + c * 32 + hi * 16;
                ka_r[c].q[0] = *(const uint4*)(pr);
                ka_r[c].q[1] = *(const uint4*)(pr + 8);
                const u16* pi = kip + c * 32 + hi * 16;
                ka_i[c].q[0] = *(const uint4*)(pi);
                ka_i[c].q[1] = *(const uint4*)(pi + 8);
            }
            v8f z = {};
            v8f srr = wmma_bf16(ka_r[1].v, qbr[1].v, wmma_bf16(ka_r[0].v, qbr[0].v, z));
            v8f sii = wmma_bf16(ka_i[1].v, qbi[1].v, wmma_bf16(ka_i[0].v, qbi[0].v, z));
            v8f sim = wmma_bf16(ka_r[1].v, qbi[1].v, wmma_bf16(ka_r[0].v, qbi[0].v,
                       wmma_bf16(ka_i[1].v, qbr[1].v, wmma_bf16(ka_i[0].v, qbr[0].v, z))));
            #pragma unroll
            for (int j = 0; j < 8; ++j) {
                p_r[t][j] = (srr[j] - sii[j]) * 0.125f;   // 1/sqrt(64)
                p_i[t][j] = sim[j] * 0.125f;
            }
        }
        // ---- streaming softmax: real ----
        {
            float tmax = p_r[0][0];
            #pragma unroll
            for (int t = 0; t < 2; ++t)
                #pragma unroll
                for (int j = 0; j < 8; ++j) tmax = fmaxf(tmax, p_r[t][j]);
            tmax = fmaxf(tmax, __shfl_xor(tmax, 16, 32));
            float mn = fmaxf(m_r, tmax);
            float alpha = __expf(m_r - mn);
            m_r = mn;
            float sum = 0.f;
            #pragma unroll
            for (int t = 0; t < 2; ++t)
                #pragma unroll
                for (int j = 0; j < 8; ++j) {
                    p_r[t][j] = __expf(p_r[t][j] - mn);
                    sum += p_r[t][j];
                }
            sum += __shfl_xor(sum, 16, 32);
            l_r = l_r * alpha + sum;
            #pragma unroll
            for (int t4 = 0; t4 < 4; ++t4) { Arr[t4] *= alpha; Ari[t4] *= alpha; }
        }
        // ---- streaming softmax: imag ----
        {
            float tmax = p_i[0][0];
            #pragma unroll
            for (int t = 0; t < 2; ++t)
                #pragma unroll
                for (int j = 0; j < 8; ++j) tmax = fmaxf(tmax, p_i[t][j]);
            tmax = fmaxf(tmax, __shfl_xor(tmax, 16, 32));
            float mn = fmaxf(m_i, tmax);
            float alpha = __expf(m_i - mn);
            m_i = mn;
            float sum = 0.f;
            #pragma unroll
            for (int t = 0; t < 2; ++t)
                #pragma unroll
                for (int j = 0; j < 8; ++j) {
                    p_i[t][j] = __expf(p_i[t][j] - mn);
                    sum += p_i[t][j];
                }
            sum += __shfl_xor(sum, 16, 32);
            l_i = l_i * alpha + sum;
            #pragma unroll
            for (int t4 = 0; t4 < 4; ++t4) { Aii[t4] *= alpha; Air[t4] *= alpha; }
        }
        // ---- pack P (Pt: 32k x 16q) into B-fragments via lane-pair swaps ----
        Frag16 pbr, pbi;
        {
            u32 r0[4], r1[4], i0[4], i1[4];
            #pragma unroll
            for (int j = 0; j < 4; ++j) {
                r0[j] = pack2(p_r[0][2 * j], p_r[0][2 * j + 1]);
                r1[j] = pack2(p_r[1][2 * j], p_r[1][2 * j + 1]);
                i0[j] = pack2(p_i[0][2 * j], p_i[0][2 * j + 1]);
                i1[j] = pack2(p_i[1][2 * j], p_i[1][2 * j + 1]);
            }
            #pragma unroll
            for (int j = 0; j < 4; ++j) {
                u32 xr1 = __shfl_xor(r1[j], 16, 32);
                u32 xr0 = __shfl_xor(r0[j], 16, 32);
                u32 xi1 = __shfl_xor(i1[j], 16, 32);
                u32 xi0 = __shfl_xor(i0[j], 16, 32);
                pbr.w[j]     = hi ? xr1 : r0[j];
                pbr.w[4 + j] = hi ? r1[j] : xr0;
                pbi.w[j]     = hi ? xi1 : i0[j];
                pbi.w[4 + j] = hi ? i1[j] : xi0;
            }
        }
        // ---- Ot[d,q] += Vt[d,k] * Pt[k,q] ----
        #pragma unroll
        for (int t4 = 0; t4 < 4; ++t4) {
            Frag16 va_r, va_i;
            const u16* pr = vbr + (t4 * 16 + ln) * VP + hi * 16;
            va_r.q[0] = *(const uint4*)(pr);
            va_r.q[1] = *(const uint4*)(pr + 8);
            const u16* pi = vbi + (t4 * 16 + ln) * VP + hi * 16;
            va_i.q[0] = *(const uint4*)(pi);
            va_i.q[1] = *(const uint4*)(pi + 8);
            Arr[t4] = wmma_bf16(va_r.v, pbr.v, Arr[t4]);
            Aii[t4] = wmma_bf16(va_i.v, pbi.v, Aii[t4]);
            Ari[t4] = wmma_bf16(va_i.v, pbr.v, Ari[t4]);
            Air[t4] = wmma_bf16(va_r.v, pbi.v, Air[t4]);
        }
        __syncthreads();
        buf ^= 1;
    }

    // ---- finalize: Or = Arr/l_r - Aii/l_i ; Oi = Ari/l_r + Air/l_i ----
    float inv_lr = 1.0f / l_r, inv_li = 1.0f / l_i;
    int b = bh >> 4, h = bh & 15;
    size_t rowO = (((size_t)(b * S_DIM + q0 + ln)) << 10) + h * 64;
    #pragma unroll
    for (int t4 = 0; t4 < 4; ++t4) {
        Out8 outr, outi;
        #pragma unroll
        for (int j = 0; j < 8; ++j) {
            outr.u[j] = f2bf(Arr[t4][j] * inv_lr - Aii[t4][j] * inv_li);
            outi.u[j] = f2bf(Ari[t4][j] * inv_lr + Air[t4][j] * inv_li);
        }
        *(uint4*)(Or_ + rowO + t4 * 16 + hi * 8) = outr.q;
        *(uint4*)(Oi_ + rowO + t4 * 16 + hi * 8) = outi.q;
    }
}

// ---------------------------------------------------------------------------
extern "C" void kernel_launch(void* const* d_in, const int* in_sizes, int n_in,
                              void* d_out, int out_size, void* d_ws, size_t ws_size,
                              hipStream_t stream) {
    (void)in_sizes; (void)n_in; (void)out_size; (void)ws_size;
    u16* ws = (u16*)d_ws;
    const size_t T  = (size_t)4096 * 1024;   // [4096,1024] bf16 tensor
    const size_t W1 = (size_t)1024 * 1024;   // [1024,1024] bf16 matrix

    u16* pin[6]; for (int i = 0; i < 6; ++i) pin[i] = ws + (size_t)i * T;
    size_t off = 6 * T;
    u16* pw[8]; for (int i = 0; i < 8; ++i) pw[i] = ws + off + (size_t)i * W1;
    off += 8 * W1;
    u16* Qr  = ws + off; off += T;  u16* Qi  = ws + off; off += T;
    u16* Kr  = ws + off; off += T;  u16* Ki  = ws + off; off += T;
    u16* Vtr = ws + off; off += T;  u16* Vti = ws + off; off += T;
    u16* Ocr = ws + off; off += T;  u16* Oci = ws + off; off += T;

    for (int i = 0; i < 6; ++i)
        pack_in_kernel<<<4096, 256, 0, stream>>>((const float*)d_in[i], pin[i]);
    const int widx[8] = {6, 7, 10, 11, 14, 15, 18, 19};
    for (int i = 0; i < 8; ++i)
        pack_w_kernel<<<1024, 256, 0, stream>>>((const float*)d_in[widx[i]], pw[i]);

    dim3 gg(32, 16);
    cgemm_kernel<0><<<gg, 256, 0, stream>>>(pin[0], pin[1], pw[0], pw[1],
        (const float*)d_in[8], (const float*)d_in[9], Qr, Qi, nullptr);
    cgemm_kernel<0><<<gg, 256, 0, stream>>>(pin[2], pin[3], pw[2], pw[3],
        (const float*)d_in[12], (const float*)d_in[13], Kr, Ki, nullptr);
    cgemm_kernel<1><<<gg, 256, 0, stream>>>(pin[4], pin[5], pw[4], pw[5],
        (const float*)d_in[16], (const float*)d_in[17], Vtr, Vti, nullptr);

    attn_kernel<<<512, 256, 0, stream>>>(Qr, Qi, Kr, Ki, Vtr, Vti, Ocr, Oci);

    cgemm_kernel<2><<<gg, 256, 0, stream>>>(Ocr, Oci, pw[6], pw[7],
        (const float*)d_in[20], (const float*)d_in[21], nullptr, nullptr, (float*)d_out);
}
